// GraphTransformerLayer_67327907332579
// MI455X (gfx1250) — matrix operand order repaired and checked by reading, not compiled, not run on hardware
//
#include <hip/hip_runtime.h>
#include <hip/hip_bf16.h>

typedef __attribute__((ext_vector_type(16))) _Float16 v16h;
typedef __attribute__((ext_vector_type(8)))  _Float16 v8h;
typedef __attribute__((ext_vector_type(8)))  float    v8f;

#define NN    50000
#define DM    256
#define NH    8
#define HD    32
#define DFF   512
#define NEG_SLOPE 0.2f
#define BN_EPS 1e-5f
#define INV_SQRT_HD 0.17677669529663687f

__device__ __forceinline__ float lrelu(float f) {
    return f >= 0.f ? f : NEG_SLOPE * f;
}
// order-preserving float<->uint encoding for atomicMax on floats
__device__ __forceinline__ unsigned encF(float f) {
    unsigned u = __float_as_uint(f);
    return (u & 0x80000000u) ? ~u : (u | 0x80000000u);
}
__device__ __forceinline__ float decF(unsigned u) {
    u = (u & 0x80000000u) ? (u & 0x7fffffffu) : ~u;
    return __uint_as_float(u);
}

// ---------------------------------------------------------------- f32 -> f16
__global__ void f32_to_f16_kernel(const float* __restrict__ in,
                                  _Float16* __restrict__ out, int n) {
    int t = blockIdx.x * blockDim.x + threadIdx.x;
    if (t < n) out[t] = (_Float16)in[t];
}

// ------------------------------------------------- pack weight W[K][N] (f32)
// into WMMA B-fragment order (f16):
// packed index p: j = p%16 (half within lane), lane = (p/16)%32,
// tile = p/512, nt = tile % (N/16), kt = tile / (N/16)
// element = W[k][n] with n = nt*16 + lane%16, k = kt*32 + (lane/16)*16 + j
__global__ void pack_w_kernel(const float* __restrict__ W,
                              _Float16* __restrict__ Wp, int K, int N) {
    int p = blockIdx.x * blockDim.x + threadIdx.x;
    if (p >= K * N) return;
    int j    = p & 15;
    int lane = (p >> 4) & 31;
    int tile = p >> 9;
    int ntiles = N >> 4;
    int nt = tile % ntiles;
    int kt = tile / ntiles;
    int n  = (nt << 4) + (lane & 15);
    int k  = (kt << 5) + ((lane >> 4) << 4) + j;
    Wp[p] = (_Float16)W[(size_t)k * N + n];
}

// ------------------------------------------------------------- WMMA GEMM
// OUT[M,N] = A[M,K](f16,row-major) @ Wp(K,N packed f16) + bias
// optional residual (f32), optional relu, f32 and/or f16 outputs.
// grid = (M/16, N/256), block = 128 (4 waves, 64 cols each)
__global__ void gemm_wmma_kernel(const _Float16* __restrict__ A,
                                 const _Float16* __restrict__ Wp,
                                 const float* __restrict__ bias,
                                 const float* __restrict__ residual,
                                 float* __restrict__ outF,
                                 _Float16* __restrict__ outH,
                                 int K, int N, int relu) {
    const int lane = threadIdx.x & 31;
    const int wave = threadIdx.x >> 5;
    const int mBase = blockIdx.x << 4;
    const int nBase = (blockIdx.y * 4 + wave) << 6;   // 64 cols per wave
    const int ntiles = N >> 4;

    v8f acc[4] = {};
    const int row  = mBase + (lane & 15);
    const int kSel = (lane >> 4) << 4;                // 0 or 16

    const _Float16* aPtr = A + (size_t)row * K + kSel;
    const _Float16* bPtr = Wp + (((size_t)(nBase >> 4)) * 32 + lane) * 16;

    const int kSteps = K >> 5;
#pragma unroll 4
    for (int kt = 0; kt < kSteps; ++kt) {
        v8h a0 = *(const v8h*)(aPtr);
        v8h a1 = *(const v8h*)(aPtr + 8);
        v16h a = __builtin_shufflevector(a0, a1, 0,1,2,3,4,5,6,7,
                                                 8,9,10,11,12,13,14,15);
        // hoist all B fragments before the WMMAs so loads batch in one clause
        v16h bfrag[4];
#pragma unroll
        for (int t = 0; t < 4; ++t)
            bfrag[t] = *(const v16h*)(bPtr + (size_t)t * 512);
#pragma unroll
        for (int t = 0; t < 4; ++t)
            acc[t] = __builtin_amdgcn_wmma_f32_16x16x32_f16(
                false, a, false, bfrag[t], (short)0, acc[t], false, false);
        aPtr += 32;
        bPtr += (size_t)ntiles * 512;
    }

    const int rbase = mBase + ((lane >> 4) << 3);
#pragma unroll
    for (int t = 0; t < 4; ++t) {
        const int col = nBase + (t << 4) + (lane & 15);
        const float bv = bias ? bias[col] : 0.f;
#pragma unroll
        for (int v = 0; v < 8; ++v) {
            const int r = rbase + v;
            float val = acc[t][v] + bv;
            if (residual) val += residual[(size_t)r * N + col];
            if (relu)     val = fmaxf(val, 0.f);
            if (outF) outF[(size_t)r * N + col] = val;
            if (outH) outH[(size_t)r * N + col] = (_Float16)val;
        }
    }
}

// -------------------------------------- per-node score + segment-max init
// s[n,h] = (Q[n,h,:].K[n,h,:]) / sqrt(HD); mEnc init with self-loop alpha
__global__ void score_init_kernel(const float* __restrict__ Q,
                                  const float* __restrict__ Kf,
                                  float* __restrict__ s,
                                  unsigned* __restrict__ mEnc) {
    int t = blockIdx.x * blockDim.x + threadIdx.x;
    if (t >= NN * NH) return;
    int n = t >> 3, h = t & 7;
    const float* q = Q  + (size_t)n * DM + h * HD;
    const float* k = Kf + (size_t)n * DM + h * HD;
    float acc = 0.f;
#pragma unroll
    for (int c = 0; c < HD; ++c) acc += q[c] * k[c];
    float sv = acc * INV_SQRT_HD;
    s[t] = sv;
    mEnc[t] = encF(lrelu(2.f * sv));   // self-loop alpha seeds the max
}

// -------------------------------------------------- edge pass 1: seg max
__global__ void edge_max_kernel(const int* __restrict__ eidx,
                                const float* __restrict__ s,
                                unsigned* __restrict__ mEnc, int E) {
    int t = blockIdx.x * blockDim.x + threadIdx.x;
    if (t >= E * NH) return;
    int e = t >> 3, h = t & 7;
    int srcN = eidx[e];
    int dstN = eidx[E + e];
    float a = lrelu(s[srcN * NH + h] + s[dstN * NH + h]);
    atomicMax(&mEnc[dstN * NH + h], encF(a));
}

// ---------------- decode max, denom = self-loop exp, overwrite mEnc with m
__global__ void self_decode_kernel(const float* __restrict__ s,
                                   unsigned* __restrict__ mEnc,
                                   float* __restrict__ denom) {
    int t = blockIdx.x * blockDim.x + threadIdx.x;
    if (t >= NN * NH) return;
    float m = decF(mEnc[t]);
    ((float*)mEnc)[t] = m;                 // reuse buffer as decoded float m
    denom[t] = __expf(lrelu(2.f * s[t]) - m);
}

// ---------------- init accumulator with self-loop term (coalesced float4)
// one thread per (node, 4-float chunk): 64 threads per node
__global__ void attn_init_kernel(const float* __restrict__ denom,
                                 const float* __restrict__ V,
                                 float* __restrict__ attn) {
    int t = blockIdx.x * blockDim.x + threadIdx.x;
    if (t >= NN * 64) return;
    int n = t >> 6, q = t & 63;
    int h = q >> 3;
    float ev = denom[n * NH + h];
    float4 v = *(const float4*)(V + (size_t)n * DM + q * 4);
    float4 o;
    o.x = ev * v.x; o.y = ev * v.y; o.z = ev * v.z; o.w = ev * v.w;
    *(float4*)(attn + (size_t)n * DM + q * 4) = o;
}

// -------------------------------------- edge pass 2: exp / denom / message
// one thread per (edge, 4-float chunk): 64 threads per edge -> coalesced
// b128 gathers of V[src] and coalesced atomic bursts on attn[dst]
__global__ void edge_acc_kernel(const int* __restrict__ eidx,
                                const float* __restrict__ s,
                                const float* __restrict__ mF,
                                float* __restrict__ denom,
                                const float* __restrict__ V,
                                float* __restrict__ attn, int E) {
    long long t = (long long)blockIdx.x * blockDim.x + threadIdx.x;
    if (t >= (long long)E * 64) return;
    int e = (int)(t >> 6);
    int q = (int)(t & 63);
    int h = q >> 3;
    int srcN = eidx[e];
    int dstN = eidx[E + e];
    float a  = lrelu(s[srcN * NH + h] + s[dstN * NH + h]);
    float ev = __expf(a - mF[dstN * NH + h]);
    if ((q & 7) == 0) atomicAdd(&denom[dstN * NH + h], ev);
    float4 v = *(const float4*)(V + (size_t)srcN * DM + q * 4);
    float* o = attn + (size_t)dstN * DM + q * 4;
    atomicAdd(o + 0, ev * v.x);
    atomicAdd(o + 1, ev * v.y);
    atomicAdd(o + 2, ev * v.z);
    atomicAdd(o + 3, ev * v.w);
}

// ------------------------------------- attn / denom -> f16 for the Wo GEMM
__global__ void attn_norm_kernel(const float* __restrict__ attn,
                                 const float* __restrict__ denom,
                                 _Float16* __restrict__ attnh) {
    int t = blockIdx.x * blockDim.x + threadIdx.x;
    if (t >= NN * DM) return;
    int n = t >> 8;
    int h = (t & 255) >> 5;
    attnh[t] = (_Float16)(attn[t] / denom[n * NH + h]);
}

// ----------------------------------------------------------- batch norm
__global__ void zero_kernel(float* __restrict__ p, int n) {
    int t = blockIdx.x * blockDim.x + threadIdx.x;
    if (t < n) p[t] = 0.f;
}

// block = NC threads, each block covers rowsPerBlock rows, atomics per block
__global__ void col_stats_kernel(const float* __restrict__ h,
                                 float* __restrict__ stats,
                                 int M, int NC, int rowsPerBlock) {
    int c  = threadIdx.x;
    int r0 = blockIdx.x * rowsPerBlock;
    float s1 = 0.f, s2 = 0.f;
    for (int i = 0; i < rowsPerBlock; ++i) {
        int r = r0 + i;
        if (r < M) {
            float v = h[(size_t)r * NC + c];
            s1 += v;
            s2 += v * v;
        }
    }
    atomicAdd(&stats[c], s1);
    atomicAdd(&stats[NC + c], s2);
}

__global__ void bn_apply_kernel(const float* __restrict__ h,
                                const float* __restrict__ stats,
                                const float* __restrict__ g,
                                const float* __restrict__ b,
                                float* __restrict__ outF,
                                _Float16* __restrict__ outH,
                                int M, int NC) {
    int t = blockIdx.x * blockDim.x + threadIdx.x;
    if (t >= M * NC) return;
    int c = t % NC;
    float mean = stats[c] / (float)M;
    float var  = stats[NC + c] / (float)M - mean * mean;
    float val  = (h[t] - mean) * rsqrtf(var + BN_EPS) * g[c] + b[c];
    if (outF) outF[t] = val;
    if (outH) outH[t] = (_Float16)val;
}

// ===========================================================================
static inline int cdiv(long long a, long long b) { return (int)((a + b - 1) / b); }

extern "C" void kernel_launch(void* const* d_in, const int* in_sizes, int n_in,
                              void* d_out, int out_size, void* d_ws, size_t ws_size,
                              hipStream_t stream) {
    const float* x     = (const float*)d_in[0];
    const int*   eidx  = (const int*)d_in[1];
    const float* Wq    = (const float*)d_in[2];
    const float* bq    = (const float*)d_in[3];
    const float* Wk    = (const float*)d_in[4];
    const float* bk    = (const float*)d_in[5];
    const float* Wv    = (const float*)d_in[6];
    const float* bv    = (const float*)d_in[7];
    const float* Wo    = (const float*)d_in[8];
    const float* bo    = (const float*)d_in[9];
    const float* W1    = (const float*)d_in[10];
    const float* b1    = (const float*)d_in[11];
    const float* W2    = (const float*)d_in[12];
    const float* b2    = (const float*)d_in[13];
    const float* bn1_g = (const float*)d_in[14];
    const float* bn1_b = (const float*)d_in[15];
    const float* bn2_g = (const float*)d_in[16];
    const float* bn2_b = (const float*)d_in[17];

    const int E = in_sizes[1] / 2;

    // -------- workspace carve-up (256B aligned bumps) --------
    char* p = (char*)d_ws;
    auto bump = [&](size_t bytes) -> void* {
        void* r = (void*)p;
        p += (bytes + 255) & ~(size_t)255;
        return r;
    };
    _Float16* xh    = (_Float16*)bump((size_t)NN * DM * 2);  // x f16; reused as hbn1 f16
    _Float16* wpq   = (_Float16*)bump((size_t)DM * DM * 2);
    _Float16* wpk   = (_Float16*)bump((size_t)DM * DM * 2);
    _Float16* wpv   = (_Float16*)bump((size_t)DM * DM * 2);
    _Float16* wpo   = (_Float16*)bump((size_t)DM * DM * 2);
    _Float16* wp1   = (_Float16*)bump((size_t)DM * DFF * 2);
    _Float16* wp2   = (_Float16*)bump((size_t)DFF * DM * 2);
    float*    Qb    = (float*)bump((size_t)NN * DM * 4);     // Q; reused: attn acc; ffn1 f16 (N*DFF*2 == N*DM*4)
    float*    Kb    = (float*)bump((size_t)NN * DM * 4);     // K; reused: h1; h2
    float*    Vb    = (float*)bump((size_t)NN * DM * 4);     // V; reused: hbn1 f32
    float*    sbuf  = (float*)bump((size_t)NN * NH * 4);
    unsigned* mbuf  = (unsigned*)bump((size_t)NN * NH * 4);  // encoded max -> decoded float m
    float*    dbuf  = (float*)bump((size_t)NN * NH * 4);     // softmax denominators
    _Float16* attnh = (_Float16*)bump((size_t)NN * DM * 2);
    float*    stats = (float*)bump((size_t)2 * DM * 4 * 2);  // room for 2*512

    float*    attn   = Qb;                                   // alias (Q dead after scores)
    float*    h1     = Kb;                                   // alias
    float*    hbn1   = Vb;                                   // alias
    _Float16* hbn1h  = xh;                                   // alias
    _Float16* ffn1h  = (_Float16*)Qb;                        // alias
    float*    h2     = Kb;                                   // alias
    float*    mF     = (float*)mbuf;

    const int TB = 256;

    // 1. x -> f16
    f32_to_f16_kernel<<<cdiv(NN * DM, TB), TB, 0, stream>>>(x, xh, NN * DM);

    // 2. pack weights into WMMA B-fragment order
    pack_w_kernel<<<cdiv(DM * DM, TB), TB, 0, stream>>>(Wq, wpq, DM, DM);
    pack_w_kernel<<<cdiv(DM * DM, TB), TB, 0, stream>>>(Wk, wpk, DM, DM);
    pack_w_kernel<<<cdiv(DM * DM, TB), TB, 0, stream>>>(Wv, wpv, DM, DM);
    pack_w_kernel<<<cdiv(DM * DM, TB), TB, 0, stream>>>(Wo, wpo, DM, DM);
    pack_w_kernel<<<cdiv(DM * DFF, TB), TB, 0, stream>>>(W1, wp1, DM, DFF);
    pack_w_kernel<<<cdiv(DFF * DM, TB), TB, 0, stream>>>(W2, wp2, DFF, DM);

    // 3. Q/K/V GEMMs (WMMA f16 -> f32)
    dim3 gg(NN / 16, DM / 256);
    gemm_wmma_kernel<<<gg, 128, 0, stream>>>(xh, wpq, bq, nullptr, Qb, nullptr, DM, DM, 0);
    gemm_wmma_kernel<<<gg, 128, 0, stream>>>(xh, wpk, bk, nullptr, Kb, nullptr, DM, DM, 0);
    gemm_wmma_kernel<<<gg, 128, 0, stream>>>(xh, wpv, bv, nullptr, Vb, nullptr, DM, DM, 0);

    // 4. per-node scores + segment softmax with self-loops
    score_init_kernel<<<cdiv(NN * NH, TB), TB, 0, stream>>>(Qb, Kb, sbuf, mbuf);
    edge_max_kernel<<<cdiv((long long)E * NH, TB), TB, 0, stream>>>(eidx, sbuf, mbuf, E);
    self_decode_kernel<<<cdiv(NN * NH, TB), TB, 0, stream>>>(sbuf, mbuf, dbuf);
    attn_init_kernel<<<cdiv((long long)NN * 64, TB), TB, 0, stream>>>(dbuf, Vb, attn);
    edge_acc_kernel<<<cdiv((long long)E * 64, TB), TB, 0, stream>>>(eidx, sbuf, mF, dbuf,
                                                                    Vb, attn, E);
    attn_norm_kernel<<<cdiv(NN * DM, TB), TB, 0, stream>>>(attn, dbuf, attnh);

    // 5. output projection + residual:  h1 = attn @ Wo + bo + x
    gemm_wmma_kernel<<<gg, 128, 0, stream>>>(attnh, wpo, bo, x, h1, nullptr, DM, DM, 0);

    // 6. BN1 -> hbn1 (f32 + f16)
    zero_kernel<<<cdiv(2 * DM, TB), TB, 0, stream>>>(stats, 2 * DM);
    col_stats_kernel<<<cdiv(NN, 100), DM, 0, stream>>>(h1, stats, NN, DM, 100);
    bn_apply_kernel<<<cdiv(NN * DM, TB), TB, 0, stream>>>(h1, stats, bn1_g, bn1_b,
                                                          hbn1, hbn1h, NN, DM);

    // 7. FFN: ffn1 = relu(hbn1 @ W1 + b1) (f16 only), h2 = ffn1 @ W2 + b2 + hbn1
    dim3 g1(NN / 16, DFF / 256);
    gemm_wmma_kernel<<<g1, 128, 0, stream>>>(hbn1h, wp1, b1, nullptr, nullptr, ffn1h,
                                             DM, DFF, 1);
    gemm_wmma_kernel<<<gg, 128, 0, stream>>>(ffn1h, wp2, b2, hbn1, h2, nullptr,
                                             DFF, DM, 0);

    // 8. BN2 -> d_out
    zero_kernel<<<cdiv(2 * DM, TB), TB, 0, stream>>>(stats, 2 * DM);
    col_stats_kernel<<<cdiv(NN, 100), DM, 0, stream>>>(h2, stats, NN, DM, 100);
    bn_apply_kernel<<<cdiv(NN * DM, TB), TB, 0, stream>>>(h2, stats, bn2_g, bn2_b,
                                                          (float*)d_out, nullptr, NN, DM);
}
